// DFL_36378372997681
// MI455X (gfx1250) — matrix operand order repaired and checked by reading, not compiled, not run on hardware
//
#include <hip/hip_runtime.h>
#include <hip/hip_bf16.h>
#include <stddef.h>

// ---------------------------------------------------------------------------
// MI455X (gfx1250). HBM-bound workload (~0.55 GB traffic vs ~310 GFLOP) ->
// implicit-GEMM conv3x3 via V_WMMA_F32_16X16X32_F16 (f16 mul, f32 acc).
// Per 32-channel K-chunk we stage ONE 10x10 halo patch of the input (covers
// all 9 taps -> 1.56x read amplification instead of 9x) and all 9 weight
// taps (contiguous 9-float runs per (co,ci)), then run 9 taps x 4 WMMAs.
// ---------------------------------------------------------------------------

typedef __attribute__((ext_vector_type(16))) _Float16 v16h;
typedef __attribute__((ext_vector_type(8)))  _Float16 v8h;
typedef __attribute__((ext_vector_type(8)))  float    v8f;

#define C_CH     256
#define HW       64
#define IMG      (C_CH * HW * HW)
#define PER_B    ((size_t)IMG)
#define WELEM    (C_CH * C_CH * 9)         // 589,824 weight elems per branch
#define NB       4
#define KCH      64
#define BATCH    16

// LDS row stride in halves: 40 halves = 80 B (16B aligned, spreads banks)
#define LROW 40

// -------------------------------- conv kernel ------------------------------
// out[b,co,h,w] = sum_{ci,kh,kw} W[co,ci,kh,kw] * in[b,ci,h+kh-1,w+kw-1]
//               + bias[co] + in[b,co,h,w]                (residual fused)
// Block: 128 threads (4 wave32 waves). Tile: 64 Cout x 64 positions (8x8).
// grid = (64 p-tiles, 4 co-tiles, 32 = which*16 + b)
__global__ __launch_bounds__(128)
void conv3x3_res_wmma(const float* __restrict__ inX, const float* __restrict__ inY,
                      const float* __restrict__ Wall, const float* __restrict__ Ball,
                      float* __restrict__ outX, float* __restrict__ outY,
                      int perBatch) {
    const int z     = blockIdx.z;
    const int which = z >> 4;
    const int b     = z & 15;

    const float* __restrict__ in  = (which ? inY  : inX) + (size_t)b * PER_B;
    float*       __restrict__ out = (which ? outY : outX) + (size_t)b * PER_B;
    const float* __restrict__ w    = Wall + (perBatch ? (size_t)b * WELEM : 0);
    const float* __restrict__ bias = Ball + (perBatch ? (size_t)b * C_CH  : 0);

    const int co0   = blockIdx.y * 64;
    const int ptile = blockIdx.x;
    const int h0    = (ptile >> 3) * 8;
    const int w0    = (ptile & 7)  * 8;

    const int tid   = threadIdx.x;
    const int lane  = tid & 31;          // wave32
    const int wave  = tid >> 5;
    const int qm    = wave >> 1;         // quadrant in M (Cout)
    const int qn    = wave & 1;          // quadrant in N (positions)
    const int col   = lane & 15;
    const int khalf = lane >> 4;

    // A: [tap 9][co 64][k 32] ; B halo: [pos 10x10 (padded 128 rows)][k 32]
    __shared__ __align__(16) _Float16 ldsA[9 * 64 * LROW];   // 46,080 B
    __shared__ __align__(16) _Float16 ldsB[128 * LROW];      // 10,240 B

    v8f acc[2][2];
#pragma unroll
    for (int mt = 0; mt < 2; ++mt)
#pragma unroll
        for (int nt = 0; nt < 2; ++nt)
#pragma unroll
            for (int r = 0; r < 8; ++r) acc[mt][nt][r] = 0.0f;

    // halo coordinates for this thread's B-staging slot (p = tid, 100 valid)
    const int prow = (tid < 100) ? (tid / 10) : 0;
    const int pcol = (tid < 100) ? (tid % 10) : 0;
    const int hh   = h0 + prow - 1;
    const int ww   = w0 + pcol - 1;
    const bool bvalid = (tid < 100) && (hh >= 0) && (hh < HW) && (ww >= 0) && (ww < HW);

    for (int kc = 0; kc < 8; ++kc) {
        const int ci0 = kc * 32;
        __syncthreads();   // LDS reuse guard

        // ---- stage A: all 9 taps, [64 co][32 ci], contiguous 9-float runs ----
        __builtin_prefetch(&w[((size_t)(co0 + (tid >> 1)) * C_CH + ((ci0 + 32) & 255)) * 9], 0, 0);
#pragma unroll 4
        for (int it = 0; it < 16; ++it) {
            const int pair = tid + it * 128;           // 2048 (co,ci) pairs
            const int co = pair >> 5, ci = pair & 31;
            const float* wp = w + ((size_t)(co0 + co) * C_CH + (ci0 + ci)) * 9;
#pragma unroll
            for (int t = 0; t < 9; ++t)
                ldsA[(t * 64 + co) * LROW + ci] = (_Float16)wp[t];
        }

        // ---- stage B: one 10x10 halo patch, [pos][32 ci], SAME padding ----
#pragma unroll 8
        for (int it = 0; it < 32; ++it) {              // ci = it, pos = tid
            float v = 0.0f;
            if (bvalid)
                v = in[(size_t)(ci0 + it) * (HW * HW) + hh * HW + ww];
            ldsB[tid * LROW + it] = (_Float16)v;       // rows 100..127 unused
        }
        __syncthreads();

        // ---- 9 taps x 4 WMMAs, fragments straight from LDS ----
#pragma unroll
        for (int tap = 0; tap < 9; ++tap) {
            const int kh = tap / 3, kw = tap % 3;      // 0..2 each

            // A 16x32 (ISA 7.12.2): lanes0-15 K=0-7,16-23 ; lanes16-31 K=8-15,24-31
            v16h afrag[2];
#pragma unroll
            for (int mt = 0; mt < 2; ++mt) {
                const _Float16* base =
                    &ldsA[(tap * 64 + qm * 32 + mt * 16 + col) * LROW];
                const v8h lo = *(const v8h*)(base + khalf * 8);
                const v8h hi = *(const v8h*)(base + 16 + khalf * 8);
#pragma unroll
                for (int j = 0; j < 8; ++j) { afrag[mt][j] = lo[j]; afrag[mt][j + 8] = hi[j]; }
            }
            // B 32x16: lanes0-15 K=0-7,16-23 ; lanes16-31 K=8-15,24-31
            v16h bfrag[2];
#pragma unroll
            for (int nt = 0; nt < 2; ++nt) {
                const int n_local = qn * 32 + nt * 16 + col;
                const int pidx = ((n_local >> 3) + kh) * 10 + (n_local & 7) + kw;
                const _Float16* base = &ldsB[pidx * LROW] + khalf * 8;
                const v8h lo = *(const v8h*)(base);
                const v8h hi = *(const v8h*)(base + 16);
#pragma unroll
                for (int j = 0; j < 8; ++j) { bfrag[nt][j] = lo[j]; bfrag[nt][j + 8] = hi[j]; }
            }

#pragma unroll
            for (int mt = 0; mt < 2; ++mt)
#pragma unroll
                for (int nt = 0; nt < 2; ++nt)
                    acc[mt][nt] = __builtin_amdgcn_wmma_f32_16x16x32_f16(
                        false, afrag[mt], false, bfrag[nt],
                        (short)0, acc[mt][nt], false, false);
        }
    }

    // ---- epilogue: C/D layout -> fused bias + residual + store ----
#pragma unroll
    for (int mt = 0; mt < 2; ++mt)
#pragma unroll
        for (int nt = 0; nt < 2; ++nt) {
            const int n_local = qn * 32 + nt * 16 + col;
            const int oh = h0 + (n_local >> 3);
            const int ow = w0 + (n_local & 7);
#pragma unroll
            for (int r = 0; r < 8; ++r) {
                const int m = co0 + qm * 32 + mt * 16 + khalf * 8 + r;
                const size_t idx = ((size_t)m * HW + oh) * HW + ow;
                out[idx] = acc[mt][nt][r] + bias[m] + in[idx];
            }
        }
}

// ------------------------- global-average-pool reduce ----------------------
__global__ __launch_bounds__(256)
void gap_sum(const float* __restrict__ x1, const float* __restrict__ y1,
             float* __restrict__ g) {
    const int bc = blockIdx.x;                     // b*256 + c
    const float* px = x1 + (size_t)bc * (HW * HW);
    const float* py = y1 + (size_t)bc * (HW * HW);
    float s = 0.0f;
    for (int i = threadIdx.x; i < HW * HW; i += 256) s += px[i] + py[i];
    __shared__ float red[256];
    red[threadIdx.x] = s;
    __syncthreads();
    for (int st = 128; st > 0; st >>= 1) {
        if (threadIdx.x < st) red[threadIdx.x] += red[threadIdx.x + st];
        __syncthreads();
    }
    if (threadIdx.x == 0) g[bc] = red[0] * (1.0f / (HW * HW));
}

// ----------------------- query MLP + softmax (per batch) -------------------
__global__ __launch_bounds__(256)
void query_mlp(const float* __restrict__ g, const float* __restrict__ scale,
               const float* __restrict__ q1w, const float* __restrict__ q1b,
               const float* __restrict__ q2w, const float* __restrict__ q2b,
               const float* __restrict__ keyw, const int* __restrict__ flag,
               float* __restrict__ att) {
    const int b = blockIdx.x;
    __shared__ float qin[258];
    __shared__ float h[512];
    __shared__ float q[KCH];
    __shared__ float logit[NB];

    for (int i = threadIdx.x; i < C_CH; i += 256) qin[i] = g[b * C_CH + i];
    if (threadIdx.x == 0) {
        const int f = flag[0];
        qin[256] = scale[0];
        qin[257] = f ? scale[1] : scale[0];
    }
    __syncthreads();

    for (int o = threadIdx.x; o < 512; o += 256) {
        float s = q1b[o];
        const float* wr = q1w + (size_t)o * 258;
        for (int i = 0; i < 258; ++i) s += qin[i] * wr[i];
        h[o] = fmaxf(s, 0.0f);
    }
    __syncthreads();

    for (int o = threadIdx.x; o < KCH; o += 256) {
        float s = q2b[o];
        const float* wr = q2w + (size_t)o * 512;
        for (int i = 0; i < 512; ++i) s += h[i] * wr[i];
        q[o] = fmaxf(s, 0.0f);
    }
    __syncthreads();

    if (threadIdx.x < NB) {
        float s = 0.0f;
        for (int i = 0; i < KCH; ++i) s += q[i] * keyw[i * NB + threadIdx.x];
        logit[threadIdx.x] = s * (1.0f / 20.0f);
    }
    __syncthreads();

    if (threadIdx.x == 0) {
        float mx = logit[0];
        for (int i = 1; i < NB; ++i) mx = fmaxf(mx, logit[i]);
        float e[NB], sum = 0.0f;
        for (int i = 0; i < NB; ++i) { e[i] = __expf(logit[i] - mx); sum += e[i]; }
        for (int i = 0; i < NB; ++i) att[b * NB + i] = e[i] / sum;
    }
}

// ----------------------- per-sample weight aggregation ---------------------
__global__ __launch_bounds__(256)
void agg_weights(const float* __restrict__ att, const float* __restrict__ mvw,
                 float* __restrict__ aggw) {
    const int b   = blockIdx.y;
    const int idx = blockIdx.x * 256 + threadIdx.x;
    if (idx >= WELEM) return;
    const float a0 = att[b * NB + 0], a1 = att[b * NB + 1];
    const float a2 = att[b * NB + 2], a3 = att[b * NB + 3];
    const float s = a0 * mvw[(size_t)0 * WELEM + idx] + a1 * mvw[(size_t)1 * WELEM + idx] +
                    a2 * mvw[(size_t)2 * WELEM + idx] + a3 * mvw[(size_t)3 * WELEM + idx];
    aggw[(size_t)b * WELEM + idx] = s;
}

__global__ __launch_bounds__(256)
void agg_bias(const float* __restrict__ att, const float* __restrict__ mvb,
              float* __restrict__ aggb) {
    const int b = blockIdx.x;
    for (int c = threadIdx.x; c < C_CH; c += 256) {
        float s = 0.0f;
        for (int nb = 0; nb < NB; ++nb) s += att[b * NB + nb] * mvb[nb * C_CH + c];
        aggb[b * C_CH + c] = s;
    }
}

// -------------------------------- launch -----------------------------------
extern "C" void kernel_launch(void* const* d_in, const int* in_sizes, int n_in,
                              void* d_out, int out_size, void* d_ws, size_t ws_size,
                              hipStream_t stream) {
    const float* x     = (const float*)d_in[0];
    const float* y     = (const float*)d_in[1];
    const float* scale = (const float*)d_in[2];
    const float* mv_w  = (const float*)d_in[3];
    const float* mv_b  = (const float*)d_in[4];
    const float* q1_w  = (const float*)d_in[5];
    const float* q1_b  = (const float*)d_in[6];
    const float* q2_w  = (const float*)d_in[7];
    const float* q2_b  = (const float*)d_in[8];
    const float* key_w = (const float*)d_in[9];
    const int*   flag  = (const int*)d_in[10];

    float* out_x = (float*)d_out;
    float* out_y = out_x + (size_t)BATCH * PER_B;

    // workspace layout (floats): x1 | y1 | g | att | aggb | aggw  (~172 MB)
    float* x1   = (float*)d_ws;
    float* y1   = x1 + (size_t)BATCH * PER_B;
    float* g    = y1 + (size_t)BATCH * PER_B;
    float* att  = g + BATCH * C_CH;
    float* aggb = att + BATCH * NB;
    float* aggw = aggb + BATCH * C_CH;

    const dim3 cgrid(64, 4, 32);   // 64 p-tiles, 4 co-tiles, 2 tensors x 16 batches
    const dim3 cblk(128);

    // 1) x1 = x + conv(x, mv_w[0]) + mv_b[0] ; y1 likewise (shared weights)
    conv3x3_res_wmma<<<cgrid, cblk, 0, stream>>>(x, y, mv_w, mv_b, x1, y1, 0);

    // 2) g[b,c] = mean(x1[b,c]) + mean(y1[b,c])
    gap_sum<<<BATCH * C_CH, 256, 0, stream>>>(x1, y1, g);

    // 3) attention weights per sample
    query_mlp<<<BATCH, 256, 0, stream>>>(g, scale, q1_w, q1_b, q2_w, q2_b,
                                         key_w, flag, att);

    // 4) per-sample aggregated conv kernels & biases
    agg_weights<<<dim3((WELEM + 255) / 256, BATCH), 256, 0, stream>>>(att, mv_w, aggw);
    agg_bias<<<BATCH, 256, 0, stream>>>(att, mv_b, aggb);

    // 5) out = dyn(x1) + x1 (per-batch weights), same for y1
    conv3x3_res_wmma<<<cgrid, cblk, 0, stream>>>(x1, y1, aggw, aggb, out_x, out_y, 1);
}
